// HybridCycle_61512521613860
// MI455X (gfx1250) — compile-verified, matrix operand order checked by reading, not run on hardware
//
#include <hip/hip_runtime.h>
#include <math.h>

// CDNA5 WMMA vector types (wave32)
typedef __attribute__((ext_vector_type(16))) _Float16 v16h;
typedef __attribute__((ext_vector_type(8)))  float    v8f;
typedef __attribute__((ext_vector_type(4)))  float    v4f;

#define B_     4
#define S_     4096
#define D_     2048
#define NP_    32
#define NM_    4
#define PROG_  32
#define NTOK   (B_ * S_)        // 16384 tokens
#define SCHUNKS 32
#define SPC    (S_ / SCHUNKS)   // 128 rows per S-chunk

// ---------------------------------------------------------------------------
// Kernel 1: column partial sums of x over S (for mean pooling).
// grid = (SCHUNKS, B, D/512), block = 256. Coalesced column walks, no atomics.
// ---------------------------------------------------------------------------
__global__ __launch_bounds__(256) void k_colsum(const float* __restrict__ x,
                                                float* __restrict__ partial) {
  const int sc = blockIdx.x;
  const int b  = blockIdx.y;
  const int dc = blockIdx.z;
  const int t  = threadIdx.x;
  for (int c = 0; c < 2; ++c) {
    const int d = dc * 512 + t + c * 256;
    const float* p = x + ((size_t)b * S_ + (size_t)sc * SPC) * D_ + d;
    float s = 0.0f;
    for (int i = 0; i < SPC; ++i) s += p[(size_t)i * D_];
    partial[((size_t)sc * B_ + b) * D_ + d] = s;
  }
}

// ---------------------------------------------------------------------------
// Kernel 2 (1 block, 256 threads): finish pooled mean, rotate pooled
// (rotation is linear so it commutes with the mean), router softmax,
// gamma/beta, per-channel rotation coefficients, f16 pre-pack of w1/w2.
// ---------------------------------------------------------------------------
__global__ __launch_bounds__(256) void k_setup(
    const float* __restrict__ partial, float* __restrict__ pooled,
    float* __restrict__ gamma, float* __restrict__ beta,
    float* __restrict__ coefA, float* __restrict__ coefB,
    int* __restrict__ partner, _Float16* __restrict__ w1h,
    _Float16* __restrict__ w2h, const float* __restrict__ angles,
    const float* __restrict__ router_w, const float* __restrict__ mode_scales,
    const float* __restrict__ mode_shifts, const float* __restrict__ w1,
    const float* __restrict__ w2, const int* __restrict__ pi,
    const int* __restrict__ pj) {
  __shared__ float logits[B_][NM_];
  __shared__ float wts[B_][NM_];
  const int t = threadIdx.x;

  // pooled mean over S
  for (int i = t; i < B_ * D_; i += 256) {
    float s = 0.0f;
    for (int sc = 0; sc < SCHUNKS; ++sc) s += partial[(size_t)sc * B_ * D_ + i];
    pooled[i] = s * (1.0f / S_);
  }
  __syncthreads();

  // rotate pooled on the 32 disjoint channel pairs
  if (t < NP_) {
    const float a = angles[t];
    const float c = cosf(a), s = sinf(a);
    const int ii = pi[t], jj = pj[t];
    for (int b = 0; b < B_; ++b) {
      const float vi = pooled[b * D_ + ii];
      const float vj = pooled[b * D_ + jj];
      pooled[b * D_ + ii] = vi * c - vj * s;
      pooled[b * D_ + jj] = vi * s + vj * c;
    }
  }
  __syncthreads();

  // router logits
  if (t < B_ * NM_) {
    const int b = t >> 2, m = t & 3;
    float s = 0.0f;
    for (int d = 0; d < D_; ++d) s += pooled[b * D_ + d] * router_w[m * D_ + d];
    logits[b][m] = s;
  }
  __syncthreads();

  // softmax over 4 modes
  if (t < B_) {
    float mx = logits[t][0];
    for (int m = 1; m < NM_; ++m) mx = fmaxf(mx, logits[t][m]);
    float e[NM_], den = 0.0f;
    for (int m = 0; m < NM_; ++m) { e[m] = expf(logits[t][m] - mx); den += e[m]; }
    for (int m = 0; m < NM_; ++m) wts[t][m] = e[m] / den;
  }
  __syncthreads();

  // gamma/beta per (batch, channel)
  for (int i = t; i < B_ * D_; i += 256) {
    const int b = i / D_, d = i % D_;
    float g = 0.0f, be = 0.0f;
    for (int m = 0; m < NM_; ++m) {
      g  += wts[b][m] * mode_scales[m * D_ + d];
      be += wts[b][m] * mode_shifts[m * D_ + d];
    }
    gamma[i] = g;
    beta[i]  = be;
  }

  // per-channel rotation coefficients: x_new[d] = coefA[d]*x[d] + coefB[d]*x[partner[d]]
  for (int d = t; d < D_; d += 256) {
    coefA[d] = 1.0f; coefB[d] = 0.0f; partner[d] = d;
  }
  __syncthreads();
  if (t < NP_) {
    const float a = angles[t];
    const float c = cosf(a), s = sinf(a);
    const int ii = pi[t], jj = pj[t];
    coefA[ii] = c;  coefB[ii] = -s; partner[ii] = jj;   // cos*xi - sin*xj
    coefA[jj] = c;  coefB[jj] =  s; partner[jj] = ii;   // sin*xi + cos*xj
  }

  // f16 weight pre-pack for WMMA B operands
  for (int i = t; i < PROG_ * D_; i += 256) w1h[i] = (_Float16)w1[i];
  for (int i = t; i < D_ * PROG_; i += 256) w2h[i] = (_Float16)w2[i];
}

// ---------------------------------------------------------------------------
// Kernel 3: streaming rotate+modulate+f16 pre-pack of x_mod.
// One thread = 8 contiguous channels (two float4 loads, one 16B half8 store).
// The rotation branch is wave-uniform except for the 64 special channels.
// ---------------------------------------------------------------------------
__global__ __launch_bounds__(256) void k_premod(
    const float* __restrict__ x, const float* __restrict__ gamma,
    const float* __restrict__ beta, const float* __restrict__ coefA,
    const float* __restrict__ coefB, const int* __restrict__ partner,
    _Float16* __restrict__ xmodh) {
  const int idx   = blockIdx.x * 256 + threadIdx.x;   // NTOK * (D/8) threads
  const int token = idx >> 8;                         // D/8 = 256 per token
  const int ch0   = (idx & 255) * 8;
  const int b     = token >> 12;                      // S = 4096
  const size_t row = (size_t)token * D_;

  const v4f x0 = *(const v4f*)(x + row + ch0);
  const v4f x1 = *(const v4f*)(x + row + ch0 + 4);
  float xv[8] = {x0.x, x0.y, x0.z, x0.w, x1.x, x1.y, x1.z, x1.w};

  _Float16 o[8];
#pragma unroll
  for (int i = 0; i < 8; ++i) {
    const int ch = ch0 + i;
    const float cb = coefB[ch];
    float v = xv[i];
    if (cb != 0.0f) v = coefA[ch] * v + cb * x[row + partner[ch]];  // rotate
    v = v * gamma[(size_t)b * D_ + ch] + beta[(size_t)b * D_ + ch]; // modulate
    o[i] = (_Float16)v;
  }
  *(ulonglong2*)(xmodh + row + ch0) = *(const ulonglong2*)o;
}

// ---------------------------------------------------------------------------
// Kernel 4: fused WMMA MLP + RMSNorm. One 128-thread block (4 waves) per
// 16-token tile.
//   GEMM1: h = gelu(x_mod @ w1^T + b1); K=2048 split across 4 waves,
//          v_wmma_f32_16x16x32_f16, cross-wave reduce in LDS.
//   GEMM2 pass1: x_gen -> silu -> d = x_out - x -> per-token sum(d^2).
//   GEMM2 pass2: recompute (1 WMMA/tile) -> RMSNorm blend -> store.
// x and x_mod are L2-resident on MI455X, so re-reads cost L2, not HBM.
// ---------------------------------------------------------------------------
__global__ __launch_bounds__(128) void k_main(
    const float* __restrict__ x, const _Float16* __restrict__ xmodh,
    const float* __restrict__ b1, const float* __restrict__ b2,
    const float* __restrict__ norm_w, const _Float16* __restrict__ w1h,
    const _Float16* __restrict__ w2h, float* __restrict__ out) {
  __shared__ float hpart[4][16][PROG_];   // per-wave partial h
  __shared__ float hfin[16][PROG_];       // reduced + gelu'd h
  __shared__ float sums_ws[4][32][8];     // d^2 partials
  __shared__ float rms_s[16];             // per-token rsqrt

  const int token0 = blockIdx.x * 16;
  const int wave   = threadIdx.x >> 5;
  const int lane   = threadIdx.x & 31;
  const int m      = lane & 15;           // row (A) / col (B) within tile
  const int khalf  = (lane >> 4) * 16;    // K sub-range held by this lane
  const size_t tokrow = (size_t)(token0 + m) * D_;

  // ---- GEMM1: each wave covers a 512-wide K slab -------------------------
  v8f acc0 = {}; v8f acc1 = {};
  const int kw0 = wave * (D_ / 4);
#pragma unroll 4
  for (int ks = 0; ks < (D_ / 4) / 32; ++ks) {      // 16 K-steps of 32
    const int kb = kw0 + ks * 32 + khalf;
    if (ks + 1 < 16) __builtin_prefetch(&xmodh[tokrow + kb + 32], 0, 0);
    const v16h a   = *(const v16h*)(xmodh + tokrow + kb);
    const v16h b0  = *(const v16h*)(w1h + (size_t)m * D_ + kb);
    const v16h b1v = *(const v16h*)(w1h + (size_t)(m + 16) * D_ + kb);
    acc0 = __builtin_amdgcn_wmma_f32_16x16x32_f16(false, a, false, b0,
                                                  (short)0, acc0, false, false);
    acc1 = __builtin_amdgcn_wmma_f32_16x16x32_f16(false, a, false, b1v,
                                                  (short)0, acc1, false, false);
  }
#pragma unroll
  for (int r = 0; r < 8; ++r) {                      // acc layout: M=r+8*hi
    const int mr = r + 8 * (lane >> 4);
    hpart[wave][mr][m]      = acc0[r];
    hpart[wave][mr][m + 16] = acc1[r];
  }
  __syncthreads();

  // cross-wave reduce, bias, exact (erf) GELU
  for (int i = threadIdx.x; i < 16 * PROG_; i += 128) {
    const int mm = i >> 5, pp = i & 31;
    const float s = hpart[0][mm][pp] + hpart[1][mm][pp] +
                    hpart[2][mm][pp] + hpart[3][mm][pp] + b1[pp];
    hfin[mm][pp] = 0.5f * s * (1.0f + erff(s * 0.70710678118654752f));
  }
  __syncthreads();

  // h as WMMA A fragment (16x32, K=32 == PROG)
  v16h ha;
#pragma unroll
  for (int i = 0; i < 16; ++i) ha[i] = (_Float16)hfin[m][khalf + i];

  // ---- GEMM2 pass 1: accumulate sum(d^2) per token -----------------------
  float dsq[8];
#pragma unroll
  for (int r = 0; r < 8; ++r) dsq[r] = 0.0f;
  for (int nt = 0; nt < 32; ++nt) {
    const int n0 = (wave * 32 + nt) * 16;
    const v16h bf = *(const v16h*)(w2h + (size_t)(n0 + m) * PROG_ + khalf);
    v8f c = {};
    c = __builtin_amdgcn_wmma_f32_16x16x32_f16(false, ha, false, bf,
                                               (short)0, c, false, false);
#pragma unroll
    for (int r = 0; r < 8; ++r) {
      const int me = r + 8 * (lane >> 4);
      const int ne = n0 + m;
      const float xg = c[r] + b2[ne];
      const float xo = xg / (1.0f + __expf(-xg));        // silu
      const float xv = x[(size_t)(token0 + me) * D_ + ne];
      const float d  = xo - xv;
      dsq[r] += d * d;
    }
  }
#pragma unroll
  for (int r = 0; r < 8; ++r) sums_ws[wave][lane][r] = dsq[r];
  __syncthreads();

  if (threadIdx.x < 16) {
    const int mm = threadIdx.x;
    const int lh = (mm >> 3) * 16;
    const int r  = mm & 7;
    float s = 0.0f;
    for (int w = 0; w < 4; ++w)
      for (int n = 0; n < 16; ++n) s += sums_ws[w][lh + n][r];
    rms_s[mm] = rsqrtf(s * (1.0f / D_) + 1.1920928955078125e-7f);
  }
  __syncthreads();

  // ---- GEMM2 pass 2: recompute, normalize, blend, store ------------------
  for (int nt = 0; nt < 32; ++nt) {
    const int n0 = (wave * 32 + nt) * 16;
    const v16h bf = *(const v16h*)(w2h + (size_t)(n0 + m) * PROG_ + khalf);
    v8f c = {};
    c = __builtin_amdgcn_wmma_f32_16x16x32_f16(false, ha, false, bf,
                                               (short)0, c, false, false);
#pragma unroll
    for (int r = 0; r < 8; ++r) {
      const int me = r + 8 * (lane >> 4);
      const int ne = n0 + m;
      const float xg = c[r] + b2[ne];
      const float xo = xg / (1.0f + __expf(-xg));
      const size_t gi = (size_t)(token0 + me) * D_ + ne;
      const float xv = x[gi];
      const float d  = xo - xv;
      out[gi] = xv + d * rms_s[me] * norm_w[ne] * 0.5f;
    }
  }
}

// ---------------------------------------------------------------------------
extern "C" void kernel_launch(void* const* d_in, const int* in_sizes, int n_in,
                              void* d_out, int out_size, void* d_ws,
                              size_t ws_size, hipStream_t stream) {
  (void)in_sizes; (void)n_in; (void)out_size; (void)ws_size;
  const float* x   = (const float*)d_in[0];
  const float* ang = (const float*)d_in[1];
  const float* rw  = (const float*)d_in[2];
  const float* msc = (const float*)d_in[3];
  const float* msh = (const float*)d_in[4];
  const float* w1  = (const float*)d_in[5];
  const float* b1  = (const float*)d_in[6];
  const float* w2  = (const float*)d_in[7];
  const float* b2  = (const float*)d_in[8];
  const float* nw  = (const float*)d_in[9];
  const int*   pi  = (const int*)d_in[10];
  const int*   pj  = (const int*)d_in[11];
  float* out = (float*)d_out;

  // workspace carve-up (~68.6 MB, dominated by the f16 x_mod tensor)
  char* ws = (char*)d_ws;
  size_t off = 0;
  float* partial = (float*)(ws + off); off += (size_t)SCHUNKS * B_ * D_ * 4;
  float* pooled  = (float*)(ws + off); off += (size_t)B_ * D_ * 4;
  float* gamma   = (float*)(ws + off); off += (size_t)B_ * D_ * 4;
  float* beta    = (float*)(ws + off); off += (size_t)B_ * D_ * 4;
  float* coefA   = (float*)(ws + off); off += (size_t)D_ * 4;
  float* coefB   = (float*)(ws + off); off += (size_t)D_ * 4;
  int*   prt     = (int*)  (ws + off); off += (size_t)D_ * 4;
  _Float16* w1h  = (_Float16*)(ws + off); off += (size_t)PROG_ * D_ * 2;
  _Float16* w2h  = (_Float16*)(ws + off); off += (size_t)D_ * PROG_ * 2;
  _Float16* xmodh = (_Float16*)(ws + off); off += (size_t)NTOK * D_ * 2;

  k_colsum<<<dim3(SCHUNKS, B_, D_ / 512), 256, 0, stream>>>(x, partial);
  k_setup<<<1, 256, 0, stream>>>(partial, pooled, gamma, beta, coefA, coefB,
                                 prt, w1h, w2h, ang, rw, msc, msh, w1, w2,
                                 pi, pj);
  k_premod<<<(NTOK * (D_ / 8)) / 256, 256, 0, stream>>>(x, gamma, beta, coefA,
                                                        coefB, prt, xmodh);
  k_main<<<NTOK / 16, 128, 0, stream>>>(x, xmodh, b1, b2, nw, w1h, w2h, out);
}
